// GraphPooling_60876866454093
// MI455X (gfx1250) — matrix operand order repaired and checked by reading, not compiled, not run on hardware
//
#include <hip/hip_runtime.h>
#include <hip/hip_bf16.h>

typedef __attribute__((ext_vector_type(16))) __bf16 v16bf;
typedef __attribute__((ext_vector_type(8)))  float  v8f;

#define D_IN     128
#define H_MID    256
#define BM       32
#define NTHREADS 128

// ---------------------------------------------------------------------------
// A-fragment loader: 16-bit A-matrix 16x32, contiguous k order (k = 16h + s), same as packed B.
// lane l holds row M = l&15; lanes<16 hold K in {0..15},
// lanes>=16 hold K in {16..31}. `row` points at A[m][k0*32].
// ---------------------------------------------------------------------------
__device__ __forceinline__ v16bf load_a_frag(const __bf16* row, int lane) {
  const int kh = (lane >> 4) * 16;
  v16bf a;
#pragma unroll
  for (int e = 0; e < 8; ++e) a[e] = row[kh + e];
#pragma unroll
  for (int e = 0; e < 8; ++e) a[e + 8] = row[kh + 8 + e];
  return a;
}

// ---------------------------------------------------------------------------
// Pack f32 row-major weight W[K][Nout] into bf16 fragment-native layout:
// Wp[((ktile*NT + ntile)*32 + lane)*16 + e] = W[ktile*32 + (lane>>4)*16 + e][ntile*16 + (lane&15)]
// so each lane's B-fragment is 16 contiguous bf16 (32B) in memory.
// ---------------------------------------------------------------------------
__global__ void pack_w_kernel(const float* __restrict__ W, __bf16* __restrict__ Wp,
                              int K, int Nout) {
  int id = blockIdx.x * blockDim.x + threadIdx.x;
  if (id >= K * Nout) return;
  int e    = id & 15;
  int lane = (id >> 4) & 31;
  int rest = id >> 9;
  int NT   = Nout >> 4;
  int ntile = rest % NT;
  int ktile = rest / NT;
  int k = ktile * 32 + (lane >> 4) * 16 + e;
  int n = ntile * 16 + (lane & 15);
  Wp[id] = (__bf16)W[(size_t)k * Nout + n];
}

// ---------------------------------------------------------------------------
// Edge phase: agg[dst] += relu(x[src] + edge_attr). One wave per edge row,
// float4 per lane (32 lanes * 4 = 128), f32 atomic adds.
// ---------------------------------------------------------------------------
__global__ void scatter_edges_kernel(const float* __restrict__ x,
                                     const int* __restrict__ src,
                                     const int* __restrict__ dst,
                                     const float* __restrict__ ea,
                                     float* __restrict__ agg, int E) {
  int e = blockIdx.x * (blockDim.x >> 5) + (threadIdx.x >> 5);
  if (e >= E) return;
  int lane = threadIdx.x & 31;
  int s = src[e];
  int d = dst[e];
  const float4 xv = *(const float4*)(x  + (size_t)s * D_IN + lane * 4);
  const float4 ev = *(const float4*)(ea + (size_t)e * D_IN + lane * 4);
  float4 m;
  m.x = fmaxf(xv.x + ev.x, 0.f);
  m.y = fmaxf(xv.y + ev.y, 0.f);
  m.z = fmaxf(xv.z + ev.z, 0.f);
  m.w = fmaxf(xv.w + ev.w, 0.f);
  float* p = agg + (size_t)d * D_IN + lane * 4;
  atomicAdd(p + 0, m.x);
  atomicAdd(p + 1, m.y);
  atomicAdd(p + 2, m.z);
  atomicAdd(p + 3, m.w);
}

// ---------------------------------------------------------------------------
// Fused GINE node phase for one conv:
//   h   = x + agg
//   t   = relu(LN_H(h @ WA + bA) * gA + beA)
//   u   = t @ WB + bB
//   out = LN_D(relu(u)) * gO + bO      -> houtF (f32, optional) + houtB (bf16)
// BM=32 rows per block, 4 waves, all GEMMs via v_wmma_f32_16x16x32_bf16.
// LayerNorm reductions: 4 threads per row, single pass (sum, sum-of-squares).
// ---------------------------------------------------------------------------
__global__ __launch_bounds__(NTHREADS) void gine_mlp_kernel(
    const float* __restrict__ xin, const float* __restrict__ agg,
    const __bf16* __restrict__ WAp, const float* __restrict__ bA,
    const float* __restrict__ gA,  const float* __restrict__ beA,
    const __bf16* __restrict__ WBp, const float* __restrict__ bB,
    const float* __restrict__ gO,  const float* __restrict__ bO,
    float* __restrict__ houtF, __bf16* __restrict__ houtB, int Nn) {
  __shared__ __bf16 As[BM][D_IN];      //  8 KB
  __shared__ float  Mid[BM][H_MID];    // 32 KB
  __shared__ __bf16 MidB[BM][H_MID];   // 16 KB
  __shared__ float  s_mean[BM], s_rstd[BM];
  __shared__ float  p1[NTHREADS], p2[NTHREADS];

  const int tid  = threadIdx.x;
  const int lane = tid & 31;
  const int wave = tid >> 5;
  const int base = blockIdx.x * BM;

  // ---- stage A = x + agg (bf16) -------------------------------------------
  for (int idx = tid; idx < BM * D_IN; idx += NTHREADS) {
    int r = idx >> 7, c = idx & 127;
    int row = base + r;
    float v = 0.f;
    if (row < Nn) v = xin[(size_t)row * D_IN + c] + agg[(size_t)row * D_IN + c];
    As[r][c] = (__bf16)v;
  }
  __syncthreads();

  // ---- GEMM1: [BM x 128] @ [128 x 256] ------------------------------------
  {
    const int NT = H_MID / 16;   // 16 n-tiles
    const int KT = D_IN / 32;    // 4 k-steps
    for (int t = wave; t < (BM / 16) * NT; t += NTHREADS / 32) {
      int mtile = t >> 4;
      int ntile = t & 15;
      v8f acc = {};
#pragma unroll
      for (int kt = 0; kt < KT; ++kt) {
        v16bf a = load_a_frag(&As[mtile * 16 + (lane & 15)][kt * 32], lane);
        v16bf b = *(const v16bf*)(WAp + (((size_t)kt * NT + ntile) * 32 + lane) * 16);
        acc = __builtin_amdgcn_wmma_f32_16x16x32_bf16(false, a, false, b,
                                                      (short)0, acc, false, false);
      }
      int n0 = ntile * 16 + (lane & 15);
      int mo = (lane >> 4) << 3;
#pragma unroll
      for (int r = 0; r < 8; ++r) Mid[mtile * 16 + mo + r][n0] = acc[r];
    }
  }
  __syncthreads();

  // ---- LayerNorm(H_MID) stats: 4 threads/row, one pass --------------------
  {
    int r = tid >> 2, sub = tid & 3;     // 64 cols per thread
    float s1 = 0.f, s2 = 0.f;
    int c0 = sub * 64;
#pragma unroll 4
    for (int c = c0; c < c0 + 64; ++c) {
      float v = Mid[r][c] + bA[c];
      s1 += v;
      s2 += v * v;
    }
    p1[tid] = s1;
    p2[tid] = s2;
  }
  __syncthreads();
  if (tid < BM) {
    float s1 = p1[tid * 4] + p1[tid * 4 + 1] + p1[tid * 4 + 2] + p1[tid * 4 + 3];
    float s2 = p2[tid * 4] + p2[tid * 4 + 1] + p2[tid * 4 + 2] + p2[tid * 4 + 3];
    float mean = s1 * (1.f / H_MID);
    float var  = s2 * (1.f / H_MID) - mean * mean;
    s_mean[tid] = mean;
    s_rstd[tid] = rsqrtf(fmaxf(var, 0.f) + 1e-5f);
  }
  __syncthreads();
  for (int idx = tid; idx < BM * H_MID; idx += NTHREADS) {
    int r = idx >> 8, c = idx & 255;
    float v = (Mid[r][c] + bA[c] - s_mean[r]) * s_rstd[r] * gA[c] + beA[c];
    MidB[r][c] = (__bf16)fmaxf(v, 0.f);
  }
  __syncthreads();

  // ---- GEMM2: [BM x 256] @ [256 x 128] -> Mid[:, 0:128] -------------------
  {
    const int NT = D_IN / 16;    // 8 n-tiles
    const int KT = H_MID / 32;   // 8 k-steps
    for (int t = wave; t < (BM / 16) * NT; t += NTHREADS / 32) {
      int mtile = t >> 3;
      int ntile = t & 7;
      v8f acc = {};
#pragma unroll
      for (int kt = 0; kt < KT; ++kt) {
        v16bf a = load_a_frag(&MidB[mtile * 16 + (lane & 15)][kt * 32], lane);
        v16bf b = *(const v16bf*)(WBp + (((size_t)kt * NT + ntile) * 32 + lane) * 16);
        acc = __builtin_amdgcn_wmma_f32_16x16x32_bf16(false, a, false, b,
                                                      (short)0, acc, false, false);
      }
      int n0 = ntile * 16 + (lane & 15);
      int mo = (lane >> 4) << 3;
#pragma unroll
      for (int r = 0; r < 8; ++r) Mid[mtile * 16 + mo + r][n0] = acc[r];
    }
  }
  __syncthreads();

  // ---- epilogue LN(D) stats over relu(u+bB): 4 threads/row, one pass ------
  {
    int r = tid >> 2, sub = tid & 3;     // 32 cols per thread
    float s1 = 0.f, s2 = 0.f;
    int c0 = sub * 32;
#pragma unroll 4
    for (int c = c0; c < c0 + 32; ++c) {
      float v = fmaxf(Mid[r][c] + bB[c], 0.f);
      s1 += v;
      s2 += v * v;
    }
    p1[tid] = s1;
    p2[tid] = s2;
  }
  __syncthreads();
  if (tid < BM) {
    float s1 = p1[tid * 4] + p1[tid * 4 + 1] + p1[tid * 4 + 2] + p1[tid * 4 + 3];
    float s2 = p2[tid * 4] + p2[tid * 4 + 1] + p2[tid * 4 + 2] + p2[tid * 4 + 3];
    float mean = s1 * (1.f / D_IN);
    float var  = s2 * (1.f / D_IN) - mean * mean;
    s_mean[tid] = mean;
    s_rstd[tid] = rsqrtf(fmaxf(var, 0.f) + 1e-5f);
  }
  __syncthreads();
  for (int idx = tid; idx < BM * D_IN; idx += NTHREADS) {
    int r = idx >> 7, c = idx & 127;
    int row = base + r;
    if (row < Nn) {
      float v = fmaxf(Mid[r][c] + bB[c], 0.f);
      float o = (v - s_mean[r]) * s_rstd[r] * gO[c] + bO[c];
      if (houtF) houtF[(size_t)row * D_IN + c] = o;
      houtB[(size_t)row * D_IN + c] = (__bf16)o;
    }
  }
}

// ---------------------------------------------------------------------------
// Final linear: out = relu(concat(h1,h2) @ wl + bl). Concat handled by
// K-splitting the A stage (k<128 -> h1, else h2).
// ---------------------------------------------------------------------------
__global__ __launch_bounds__(NTHREADS) void final_linear_kernel(
    const __bf16* __restrict__ h1b, const __bf16* __restrict__ h2b,
    const __bf16* __restrict__ WLp, const float* __restrict__ bl,
    float* __restrict__ out, int Nn) {
  __shared__ __bf16 As[BM][256];   // 16 KB
  const int tid  = threadIdx.x;
  const int lane = tid & 31;
  const int wave = tid >> 5;
  const int base = blockIdx.x * BM;

  for (int idx = tid; idx < BM * 256; idx += NTHREADS) {
    int r = idx >> 8, c = idx & 255;
    int row = base + r;
    __bf16 v = (__bf16)0.f;
    if (row < Nn)
      v = (c < 128) ? h1b[(size_t)row * 128 + c] : h2b[(size_t)row * 128 + (c - 128)];
    As[r][c] = v;
  }
  __syncthreads();

  const int NT = 8, KT = 8;
  for (int t = wave; t < 2 * NT; t += NTHREADS / 32) {
    int mtile = t >> 3;
    int ntile = t & 7;
    v8f acc = {};
#pragma unroll
    for (int kt = 0; kt < KT; ++kt) {
      v16bf a = load_a_frag(&As[mtile * 16 + (lane & 15)][kt * 32], lane);
      v16bf b = *(const v16bf*)(WLp + (((size_t)kt * NT + ntile) * 32 + lane) * 16);
      acc = __builtin_amdgcn_wmma_f32_16x16x32_bf16(false, a, false, b,
                                                    (short)0, acc, false, false);
    }
    int n0 = ntile * 16 + (lane & 15);
    int mo = (lane >> 4) << 3;
#pragma unroll
    for (int r = 0; r < 8; ++r) {
      int row = base + mtile * 16 + mo + r;
      if (row < Nn) out[(size_t)row * 128 + n0] = fmaxf(acc[r] + bl[n0], 0.f);
    }
  }
}

// ---------------------------------------------------------------------------
extern "C" void kernel_launch(void* const* d_in, const int* in_sizes, int n_in,
                              void* d_out, int out_size, void* d_ws, size_t ws_size,
                              hipStream_t stream) {
  const float* x    = (const float*)d_in[0];
  const int*   eidx = (const int*)d_in[1];
  const float* ea   = (const float*)d_in[2];
  // d_in[3] = pos (unused by reference)
  const float* w1a  = (const float*)d_in[4];
  const float* b1a  = (const float*)d_in[5];
  const float* g1a  = (const float*)d_in[6];
  const float* be1a = (const float*)d_in[7];
  const float* w1b  = (const float*)d_in[8];
  const float* b1b  = (const float*)d_in[9];
  const float* w2a  = (const float*)d_in[10];
  const float* b2a  = (const float*)d_in[11];
  const float* g2a  = (const float*)d_in[12];
  const float* be2a = (const float*)d_in[13];
  const float* w2b  = (const float*)d_in[14];
  const float* b2b  = (const float*)d_in[15];
  const float* ng1  = (const float*)d_in[16];
  const float* nb1  = (const float*)d_in[17];
  const float* ng2  = (const float*)d_in[18];
  const float* nb2  = (const float*)d_in[19];
  const float* wl   = (const float*)d_in[20];
  const float* bl   = (const float*)d_in[21];

  const int N = in_sizes[0] / D_IN;
  const int E = in_sizes[1] / 2;
  const int* src = eidx;
  const int* dst = eidx + E;

  // workspace carve-up (256B aligned)
  char* ws = (char*)d_ws;
  size_t off = 0;
  auto alloc = [&](size_t bytes) -> char* {
    char* p = ws + off;
    off += (bytes + 255) & ~(size_t)255;
    return p;
  };
  float*  agg  = (float*)alloc((size_t)N * D_IN * sizeof(float));
  float*  h1f  = (float*)alloc((size_t)N * D_IN * sizeof(float));
  __bf16* h1b  = (__bf16*)alloc((size_t)N * D_IN * sizeof(__bf16));
  __bf16* h2b  = (__bf16*)alloc((size_t)N * D_IN * sizeof(__bf16));
  __bf16* w1ap = (__bf16*)alloc((size_t)D_IN * H_MID * sizeof(__bf16));
  __bf16* w1bp = (__bf16*)alloc((size_t)H_MID * D_IN * sizeof(__bf16));
  __bf16* w2ap = (__bf16*)alloc((size_t)D_IN * H_MID * sizeof(__bf16));
  __bf16* w2bp = (__bf16*)alloc((size_t)H_MID * D_IN * sizeof(__bf16));
  __bf16* wlp  = (__bf16*)alloc((size_t)H_MID * D_IN * sizeof(__bf16));
  (void)ws_size; (void)n_in; (void)out_size;

  // pack all weights into WMMA fragment-native bf16 layout
  {
    const int n = D_IN * H_MID;   // 32768 elements each
    const int blk = 256, grd = (n + blk - 1) / blk;
    pack_w_kernel<<<grd, blk, 0, stream>>>(w1a, w1ap, D_IN, H_MID);
    pack_w_kernel<<<grd, blk, 0, stream>>>(w1b, w1bp, H_MID, D_IN);
    pack_w_kernel<<<grd, blk, 0, stream>>>(w2a, w2ap, D_IN, H_MID);
    pack_w_kernel<<<grd, blk, 0, stream>>>(w2b, w2bp, H_MID, D_IN);
    pack_w_kernel<<<grd, blk, 0, stream>>>(wl,  wlp,  H_MID, D_IN);
  }

  const int nodeBlocks = (N + BM - 1) / BM;
  const int edgeBlocks = (E + 7) / 8;   // 8 waves (edges) per 256-thread block

  // ---- conv1 ----
  hipMemsetAsync(agg, 0, (size_t)N * D_IN * sizeof(float), stream);
  scatter_edges_kernel<<<edgeBlocks, 256, 0, stream>>>(x, src, dst, ea, agg, E);
  gine_mlp_kernel<<<nodeBlocks, NTHREADS, 0, stream>>>(
      x, agg, w1ap, b1a, g1a, be1a, w1bp, b1b, ng1, nb1, h1f, h1b, N);

  // ---- conv2 ----
  hipMemsetAsync(agg, 0, (size_t)N * D_IN * sizeof(float), stream);
  scatter_edges_kernel<<<edgeBlocks, 256, 0, stream>>>(h1f, src, dst, ea, agg, E);
  gine_mlp_kernel<<<nodeBlocks, NTHREADS, 0, stream>>>(
      h1f, agg, w2ap, b2a, g2a, be2a, w2bp, b2b, ng2, nb2, nullptr, h2b, N);

  // ---- final linear on concat ----
  final_linear_kernel<<<nodeBlocks, NTHREADS, 0, stream>>>(
      h1b, h2b, wlp, bl, (float*)d_out, N);
}
